// SparseMoE_33947421508244
// MI455X (gfx1250) — matrix operand order repaired and checked by reading, not compiled, not run on hardware
//
#include <hip/hip_runtime.h>
#include <math.h>

#define NTOK 8192
#define DDIM 1024
#define FDIM 1024
#define NEXP 8

typedef __attribute__((ext_vector_type(16))) __bf16 v16bf;
typedef __attribute__((ext_vector_type(8)))  float  v8f;
typedef __attribute__((ext_vector_type(4)))  unsigned int u32x4;
typedef __attribute__((ext_vector_type(4)))  float  f32x4;
typedef __attribute__((ext_vector_type(4)))  __bf16 bf16x4;

union Frag32 { v16bf bf; u32x4 q[2]; };

// ---------------- zero output + expert counters ----------------
__global__ void zero_kernel(float* __restrict__ out, int* __restrict__ cnt) {
    size_t i = (size_t)blockIdx.x * blockDim.x + threadIdx.x;
    f32x4 z = {0.f, 0.f, 0.f, 0.f};
    if (i < (size_t)NTOK * FDIM / 4) ((f32x4*)out)[i] = z;
    if (i < NEXP) cnt[i] = 0;
}

// ---------------- We [E][D][F] f32 -> Wet [E][F][D] bf16 (tiled transpose) ---
__global__ __launch_bounds__(256) void transpose_we_kernel(
    const float* __restrict__ We, __bf16* __restrict__ Wet) {
    __shared__ float tile[32][33];
    int bid = blockIdx.x;                 // E * (D/32) * (F/32) = 8*32*32
    int e  = bid >> 10;
    int r  = bid & 1023;
    int dt = r >> 5;
    int ft = r & 31;
    int tid = threadIdx.x;
    int col = tid & 31;
    const float* src = We + ((size_t)e * DDIM + dt * 32) * FDIM + ft * 32;
    for (int i = tid >> 5; i < 32; i += 8)
        tile[i][col] = src[(size_t)i * FDIM + col];
    __syncthreads();
    __bf16* dst = Wet + ((size_t)e * FDIM + ft * 32) * DDIM + dt * 32;
    for (int i = tid >> 5; i < 32; i += 8)
        dst[(size_t)i * DDIM + col] = (__bf16)tile[col][i];
}

// ---------------- router: logits, top-2 softmax, x->bf16, expert lists ------
__global__ __launch_bounds__(256) void router_kernel(
    const float* __restrict__ x, const float* __restrict__ Wr,
    const float* __restrict__ br, __bf16* __restrict__ xb,
    int* __restrict__ cnt, int* __restrict__ tokList, float* __restrict__ wList) {
    __shared__ float Wrs[DDIM * NEXP];    // 32 KB
    int tid = threadIdx.x;
    for (int i = tid; i < DDIM * NEXP; i += 256) Wrs[i] = Wr[i];
    __syncthreads();

    int n = blockIdx.x * 256 + tid;       // grid = NTOK/256
    float acc[NEXP];
#pragma unroll
    for (int e = 0; e < NEXP; ++e) acc[e] = br[e];

    const float* xr = x + (size_t)n * DDIM;
    __bf16* xbr = xb + (size_t)n * DDIM;
    for (int d = 0; d < DDIM; d += 4) {
        f32x4 xv = *(const f32x4*)(xr + d);
        bf16x4 pk;
#pragma unroll
        for (int q = 0; q < 4; ++q) pk[q] = (__bf16)xv[q];
        *(bf16x4*)(xbr + d) = pk;
#pragma unroll
        for (int q = 0; q < 4; ++q) {
            float xs = xv[q];
#pragma unroll
            for (int e = 0; e < NEXP; ++e)
                acc[e] = fmaf(xs, Wrs[(d + q) * NEXP + e], acc[e]);
        }
    }

    // top-2 (stable: first index wins ties, matching lax.top_k)
    int b0 = 0; float v0 = acc[0];
#pragma unroll
    for (int e = 1; e < NEXP; ++e) if (acc[e] > v0) { v0 = acc[e]; b0 = e; }
    int b1 = -1; float v1 = -3.4e38f;
#pragma unroll
    for (int e = 0; e < NEXP; ++e)
        if (e != b0 && acc[e] > v1) { v1 = acc[e]; b1 = e; }

    float e1 = expf(v1 - v0);
    float inv = 1.f / (1.f + e1);
    float w0 = inv, w1 = e1 * inv;

    int p0 = atomicAdd(&cnt[b0], 1);
    tokList[b0 * NTOK + p0] = n;  wList[b0 * NTOK + p0] = w0;
    int p1 = atomicAdd(&cnt[b1], 1);
    tokList[b1 * NTOK + p1] = n;  wList[b1 * NTOK + p1] = w1;
}

// ---------------- grouped expert GEMM: 128x128 tile, 8 waves, bf16 WMMA -----
__global__ __launch_bounds__(256) void moe_gemm_kernel(
    const __bf16* __restrict__ xb, const __bf16* __restrict__ wet,
    const float* __restrict__ be, const int* __restrict__ cnt,
    const int* __restrict__ tokList, const float* __restrict__ wList,
    float* __restrict__ out) {
    const int MTILES = NTOK / 128;                 // 64
    int e  = blockIdx.x / MTILES;
    int mt = blockIdx.x % MTILES;
    int ce = cnt[e];
    int m0 = mt * 128;
    if (m0 >= ce) return;                          // uniform early exit
    int rows = ce - m0; if (rows > 128) rows = 128;
    int fbase = blockIdx.y * 128;

    __shared__ int   toks[128];
    __shared__ float wts[128];
    int tid = threadIdx.x;
    if (tid < 128) {
        bool valid = tid < rows;
        toks[tid] = valid ? tokList[e * NTOK + m0 + tid] : 0;
        wts[tid]  = valid ? wList[e * NTOK + m0 + tid] : 0.f;
    }
    __syncthreads();

    int lane = tid & 31;
    int w    = tid >> 5;
    int wm   = w & 1;          // M offset wm*64  (2 wave-rows)
    int wf   = w >> 1;         // F offset wf*32  (4 wave-cols)
    int ln   = lane & 15;
    int lh   = lane >> 4;      // 0/1 half-wave

    const __bf16* aBase[4];
#pragma unroll
    for (int i = 0; i < 4; ++i) {
        int r = wm * 64 + i * 16 + ln;
        aBase[i] = xb + (size_t)toks[r] * DDIM;
    }
    const __bf16* bBase[2];
#pragma unroll
    for (int j = 0; j < 2; ++j) {
        int f = fbase + wf * 32 + j * 16 + ln;
        bBase[j] = wet + ((size_t)e * FDIM + f) * DDIM;
    }

    v8f c[4][2];
#pragma unroll
    for (int i = 0; i < 4; ++i)
#pragma unroll
        for (int j = 0; j < 2; ++j)
            c[i][j] = (v8f){0.f, 0.f, 0.f, 0.f, 0.f, 0.f, 0.f, 0.f};

    int ka = lh * 8;           // A: per-lane K chunks at khalf*8 and 16+khalf*8
    int kb = lh * 8;           // B: per-lane K chunks at khalf*8 and 16+khalf*8

    for (int k = 0; k < DDIM; k += 32) {
        Frag32 a[4], b[2];
#pragma unroll
        for (int i = 0; i < 4; ++i) {
            a[i].q[0] = *(const u32x4*)(aBase[i] + k + ka);
            a[i].q[1] = *(const u32x4*)(aBase[i] + k + 16 + ka);
        }
#pragma unroll
        for (int j = 0; j < 2; ++j) {
            b[j].q[0] = *(const u32x4*)(bBase[j] + k + kb);
            b[j].q[1] = *(const u32x4*)(bBase[j] + k + 16 + kb);
        }
#pragma unroll
        for (int i = 0; i < 4; ++i)
#pragma unroll
            for (int j = 0; j < 2; ++j)
                c[i][j] = __builtin_amdgcn_wmma_f32_16x16x32_bf16(
                    false, a[i].bf, false, b[j].bf, (short)0, c[i][j],
                    false, false);
    }

    // epilogue: bias + exact GELU + gate weight + atomic combine (2 adds/elem)
#pragma unroll
    for (int j = 0; j < 2; ++j) {
        int f = fbase + wf * 32 + j * 16 + ln;
        float bias = be[e * FDIM + f];
#pragma unroll
        for (int i = 0; i < 4; ++i) {
#pragma unroll
            for (int v = 0; v < 8; ++v) {
                int m = wm * 64 + i * 16 + lh * 8 + v;
                if (m < rows) {
                    float h = c[i][j][v] + bias;
                    float g = 0.5f * h * (1.0f + erff(h * 0.70710678118654752f));
                    atomicAdd(out + (size_t)toks[m] * FDIM + f, wts[m] * g);
                }
            }
        }
    }
}

// ---------------- launch ----------------------------------------------------
extern "C" void kernel_launch(void* const* d_in, const int* in_sizes, int n_in,
                              void* d_out, int out_size, void* d_ws, size_t ws_size,
                              hipStream_t stream) {
    const float* x  = (const float*)d_in[0];
    const float* Wr = (const float*)d_in[1];
    const float* br = (const float*)d_in[2];
    const float* We = (const float*)d_in[3];
    const float* be = (const float*)d_in[4];
    float* out = (float*)d_out;

    char* ws = (char*)d_ws;
    __bf16* xb  = (__bf16*)(ws);                                   // 16 MB
    __bf16* wet = (__bf16*)(ws + (size_t)16 * 1024 * 1024);        // 16 MB
    char* tail  = ws + (size_t)32 * 1024 * 1024;
    int*   cnt     = (int*)tail;                                   // 32 B
    int*   tokList = (int*)(tail + 4096);                          // 256 KB
    float* wList   = (float*)(tail + 4096 + sizeof(int) * NEXP * NTOK);

    zero_kernel<<<(NTOK * FDIM / 4 + 255) / 256, 256, 0, stream>>>(out, cnt);
    transpose_we_kernel<<<NEXP * (DDIM / 32) * (FDIM / 32), 256, 0, stream>>>(We, wet);
    router_kernel<<<NTOK / 256, 256, 0, stream>>>(x, Wr, br, xb, cnt, tokList, wList);

    dim3 grid(NEXP * (NTOK / 128), FDIM / 128);   // (512, 8)
    moe_gemm_kernel<<<grid, 256, 0, stream>>>(xb, wet, be, cnt, tokList, wList, out);
}
